// Decoder_48696339202140
// MI455X (gfx1250) — matrix operand order repaired and checked by reading, not compiled, not run on hardware
//
#include <hip/hip_runtime.h>
#include <hip/hip_bf16.h>
#include <cstdint>

// Problem dims (fixed by the reference)
#define B_   64
#define S_   20
#define V_   10000
#define E_   512
#define L_   512
#define A_   512
#define D_   2048
#define P_   196          // 14*14
#define T_   19           // S-1
#define KXH  3072         // E + D + L  (fused [x|h] GEMM K)
#define NCLS_PAD 10048    // V padded up to a multiple of 64

typedef __attribute__((ext_vector_type(16))) _Float16 v16h;
typedef __attribute__((ext_vector_type(8)))  _Float16 v8h;
typedef __attribute__((ext_vector_type(8)))  float    v8f;

static __device__ __forceinline__ v16h cat8(v8h lo, v8h hi) {
  return __builtin_shufflevector(lo, hi, 0,1,2,3,4,5,6,7,8,9,10,11,12,13,14,15);
}

// ---------------------------------------------------------------------------
// WMMA GEMM:  C[M,N] = A[M,K] * B[N,K]^T + bias[N]
// block = 128 threads (4 waves); each wave owns ONE 16-wide N tile and FOUR
// 16-tall M tiles (4 accumulators). The B (weight) operand is loaded once per
// K-step and reused by 4 v_wmma ops -> 4x less weight traffic from L2, which
// is the dominant cost for the M=64 recurrent GEMMs.
// grid = (N/64, M/64).  K % 32 == 0, N % 64 == 0, M % 64 == 0.
// ---------------------------------------------------------------------------
template <bool A_IS_F32>
__global__ void wmma_gemm_kernel(const void* __restrict__ Aptr,
                                 const _Float16* __restrict__ Bw,
                                 const float* __restrict__ bias,
                                 float* __restrict__ Cout,
                                 _Float16* __restrict__ Cout16,
                                 int M, int N, int K) {
  const int wave = threadIdx.x >> 5;
  const int lane = threadIdx.x & 31;
  const int half = lane >> 4;      // K-half selector for A/B operands
  const int mr   = lane & 15;      // A row within tile == C column within tile
  const int n0   = (blockIdx.x * 4 + wave) * 16;
  const int m0   = blockIdx.y * 64;
  (void)M;

  v8f acc0 = {0.f,0.f,0.f,0.f,0.f,0.f,0.f,0.f};
  v8f acc1 = acc0, acc2 = acc0, acc3 = acc0;

  const _Float16* Brow = Bw + (size_t)(n0 + mr) * K + half * 8;
  const size_t arow0 = (size_t)(m0 +  0 + mr) * K;
  const size_t arow1 = (size_t)(m0 + 16 + mr) * K;
  const size_t arow2 = (size_t)(m0 + 32 + mr) * K;
  const size_t arow3 = (size_t)(m0 + 48 + mr) * K;

  for (int k = 0; k < K; k += 32) {
    v8h b0 = *(const v8h*)(Brow + k);
    v8h b1 = *(const v8h*)(Brow + k + 16);
    v16h bm = cat8(b0, b1);

    auto loadA = [&](size_t arow) -> v16h {
      if (A_IS_F32) {
        const float* Af = (const float*)Aptr;
        v8f a0 = *(const v8f*)(Af + arow + k + half * 8);
        v8f a1 = *(const v8f*)(Af + arow + k + 16 + half * 8);
        v16h a;
#pragma unroll
        for (int i = 0; i < 8; ++i) { a[i] = (_Float16)a0[i]; a[i + 8] = (_Float16)a1[i]; }
        return a;
      } else {
        const _Float16* Ah = (const _Float16*)Aptr;
        v8h a0 = *(const v8h*)(Ah + arow + k + half * 8);
        v8h a1 = *(const v8h*)(Ah + arow + k + 16 + half * 8);
        return cat8(a0, a1);
      }
    };

    acc0 = __builtin_amdgcn_wmma_f32_16x16x32_f16(false, loadA(arow0), false, bm,
                                                  (short)0, acc0, false, false);
    acc1 = __builtin_amdgcn_wmma_f32_16x16x32_f16(false, loadA(arow1), false, bm,
                                                  (short)0, acc1, false, false);
    acc2 = __builtin_amdgcn_wmma_f32_16x16x32_f16(false, loadA(arow2), false, bm,
                                                  (short)0, acc2, false, false);
    acc3 = __builtin_amdgcn_wmma_f32_16x16x32_f16(false, loadA(arow3), false, bm,
                                                  (short)0, acc3, false, false);
  }

  const int col = n0 + mr;
  const float bv = bias ? bias[col] : 0.f;
#define EPILOGUE(ACC, MI)                                                     \
  _Pragma("unroll")                                                           \
  for (int r = 0; r < 8; ++r) {                                               \
    const int m = m0 + (MI)*16 + half * 8 + r;                                \
    const float v = (ACC)[r] + bv;                                            \
    Cout[(size_t)m * N + col] = v;                                            \
    if (Cout16) Cout16[(size_t)m * N + col] = (_Float16)v;                    \
  }
  EPILOGUE(acc0, 0) EPILOGUE(acc1, 1) EPILOGUE(acc2, 2) EPILOGUE(acc3, 3)
#undef EPILOGUE
}

// ---------------------------------------------------------------------------
// Classifier GEMM with fused sequence mask:  d_out[b,t,:] = mask ? h@Wcls^T+b : 0
// Same 4-M-tile blocking: one wave covers all 64 batch rows, so each W_cls
// element is pulled from L2 exactly once per timestep.  grid = (NCLS_PAD/64, 1).
// ---------------------------------------------------------------------------
__global__ void wmma_cls_kernel(const _Float16* __restrict__ H16,   // [B, L]
                                const _Float16* __restrict__ Wc,    // [NCLS_PAD, L]
                                const float* __restrict__ bcls,     // [V]
                                const int* __restrict__ lengths,    // [B]
                                float* __restrict__ Yout,           // [B, T, V]
                                int t) {
  const int wave = threadIdx.x >> 5;
  const int lane = threadIdx.x & 31;
  const int half = lane >> 4;
  const int mr   = lane & 15;
  const int n0   = (blockIdx.x * 4 + wave) * 16;

  v8f acc0 = {0.f,0.f,0.f,0.f,0.f,0.f,0.f,0.f};
  v8f acc1 = acc0, acc2 = acc0, acc3 = acc0;

  const _Float16* Brow = Wc + (size_t)(n0 + mr) * L_ + half * 8;
  const size_t arow0 = (size_t)( 0 + mr) * L_;
  const size_t arow1 = (size_t)(16 + mr) * L_;
  const size_t arow2 = (size_t)(32 + mr) * L_;
  const size_t arow3 = (size_t)(48 + mr) * L_;

  for (int k = 0; k < L_; k += 32) {
    v8h b0 = *(const v8h*)(Brow + k);
    v8h b1 = *(const v8h*)(Brow + k + 16);
    v16h bm = cat8(b0, b1);
    auto loadA = [&](size_t arow) -> v16h {
      v8h a0 = *(const v8h*)(H16 + arow + k + half * 8);
      v8h a1 = *(const v8h*)(H16 + arow + k + 16 + half * 8);
      return cat8(a0, a1);
    };
    acc0 = __builtin_amdgcn_wmma_f32_16x16x32_f16(false, loadA(arow0), false, bm,
                                                  (short)0, acc0, false, false);
    acc1 = __builtin_amdgcn_wmma_f32_16x16x32_f16(false, loadA(arow1), false, bm,
                                                  (short)0, acc1, false, false);
    acc2 = __builtin_amdgcn_wmma_f32_16x16x32_f16(false, loadA(arow2), false, bm,
                                                  (short)0, acc2, false, false);
    acc3 = __builtin_amdgcn_wmma_f32_16x16x32_f16(false, loadA(arow3), false, bm,
                                                  (short)0, acc3, false, false);
  }

  const int col = n0 + mr;
  if (col >= V_) return;
  const float bv = bcls[col];
#define EPILOGUE(ACC, MI)                                                     \
  _Pragma("unroll")                                                           \
  for (int r = 0; r < 8; ++r) {                                               \
    const int b = (MI)*16 + half * 8 + r;                                     \
    const bool msk = t < (lengths[b] - 1);                                    \
    Yout[((size_t)b * T_ + t) * V_ + col] = msk ? ((ACC)[r] + bv) : 0.f;      \
  }
  EPILOGUE(acc0, 0) EPILOGUE(acc1, 1) EPILOGUE(acc2, 2) EPILOGUE(acc3, 3)
#undef EPILOGUE
}

// ---------------------------------------------------------------------------
// Attention: e = relu(att1 + att2) @ W_att + b_att ; softmax over P ; awe.
// One block per batch element, fp32 throughout (precision-sensitive, tiny).
// ---------------------------------------------------------------------------
__global__ void attn_kernel(const float* __restrict__ att1,   // [B*P, A]
                            const float* __restrict__ att2,   // [B, A]
                            const float* __restrict__ Watt,   // [A]
                            const float* __restrict__ batt,   // [1]
                            const float* __restrict__ feats,  // [B, P, D] fp32
                            const int* __restrict__ lengths,
                            float* __restrict__ awe,          // [B, D]
                            float* __restrict__ alph_out,     // [B, T, P]
                            int t) {
  __shared__ float s_a2[A_];
  __shared__ float s_w[A_];
  __shared__ float s_al[256];
  __shared__ float s_red[256];
  const int b = blockIdx.x, tid = threadIdx.x;

  for (int i = tid; i < A_; i += 256) { s_a2[i] = att2[b * A_ + i]; s_w[i] = Watt[i]; }
  __syncthreads();

  float e = -1e30f;
  if (tid < P_) {
    const float* row = att1 + ((size_t)b * P_ + tid) * A_;
    float acc = batt[0];
    for (int a = 0; a < A_; ++a) {
      const float v = row[a] + s_a2[a];
      acc += (v > 0.f ? v : 0.f) * s_w[a];
    }
    e = acc;
  }
  s_red[tid] = e; __syncthreads();
  for (int s = 128; s > 0; s >>= 1) {
    if (tid < s) s_red[tid] = fmaxf(s_red[tid], s_red[tid + s]);
    __syncthreads();
  }
  const float mx = s_red[0]; __syncthreads();
  const float ex = (tid < P_) ? __expf(e - mx) : 0.f;
  s_red[tid] = ex; __syncthreads();
  for (int s = 128; s > 0; s >>= 1) {
    if (tid < s) s_red[tid] += s_red[tid + s];
    __syncthreads();
  }
  const float al = ex / s_red[0];
  s_al[tid] = al;
  if (tid < P_) {
    const bool msk = t < (lengths[b] - 1);
    alph_out[((size_t)b * T_ + t) * P_ + tid] = msk ? al : 0.f;
  }
  __syncthreads();

  for (int d = tid; d < D_; d += 256) {
    float acc = 0.f;
    const float* fb = feats + (size_t)b * P_ * D_ + d;
    for (int p = 0; p < P_; ++p) acc += s_al[p] * fb[(size_t)p * D_];
    awe[b * D_ + d] = acc;
  }
}

// ---------------------------------------------------------------------------
// Small setup / pointwise kernels
// ---------------------------------------------------------------------------
__global__ void mean_kernel(const float* __restrict__ feats, float* __restrict__ meanf) {
  const int b = blockIdx.x;
  const int d = blockIdx.y * 256 + threadIdx.x;
  const float* fb = feats + (size_t)b * P_ * D_ + d;
  float acc = 0.f;
  for (int p = 0; p < P_; ++p) acc += fb[(size_t)p * D_];
  meanf[b * D_ + d] = acc * (1.f / (float)P_);
}

__global__ void h0c0_kernel(const float* __restrict__ meanf,
                            const float* __restrict__ Wh, const float* __restrict__ bh,
                            const float* __restrict__ Wc, const float* __restrict__ bc,
                            float* __restrict__ h, float* __restrict__ c,
                            _Float16* __restrict__ h16) {
  const int b = blockIdx.x;
  const int j = blockIdx.y * 256 + threadIdx.x;
  const float* mf = meanf + b * D_;
  const float* wh = Wh + (size_t)j * D_;
  const float* wc = Wc + (size_t)j * D_;
  float ha = bh[j], ca = bc[j];
  for (int d = 0; d < D_; ++d) { const float m = mf[d]; ha += m * wh[d]; ca += m * wc[d]; }
  h[b * L_ + j] = ha; c[b * L_ + j] = ca; h16[b * L_ + j] = (_Float16)ha;
}

__global__ void cvt_f16_kernel(const float* __restrict__ in, _Float16* __restrict__ out,
                               size_t n) {
  const size_t i = (size_t)blockIdx.x * 256 + threadIdx.x;
  if (i < n) out[i] = (_Float16)in[i];
}

__global__ void cvt_cls_kernel(const float* __restrict__ Wcls, _Float16* __restrict__ out) {
  const size_t i = (size_t)blockIdx.x * 256 + threadIdx.x;
  if (i >= (size_t)NCLS_PAD * L_) return;
  const size_t nrow = i / L_;
  out[i] = (nrow < (size_t)V_) ? (_Float16)Wcls[i] : (_Float16)0.f;
}

__global__ void build_wg_kernel(const float* __restrict__ Wih, const float* __restrict__ Whh,
                                _Float16* __restrict__ Wg) {
  const size_t i = (size_t)blockIdx.x * 256 + threadIdx.x;
  if (i >= (size_t)(4 * L_) * KXH) return;
  const size_t row = i / KXH, col = i % KXH;
  const float v = (col < (size_t)(E_ + D_)) ? Wih[row * (E_ + D_) + col]
                                            : Whh[row * L_ + (col - (E_ + D_))];
  Wg[i] = (_Float16)v;
}

__global__ void bias_sum_kernel(const float* __restrict__ bih, const float* __restrict__ bhh,
                                float* __restrict__ bg) {
  const int i = blockIdx.x * 256 + threadIdx.x;
  if (i < 4 * L_) bg[i] = bih[i] + bhh[i];
}

__global__ void embed_kernel(const int* __restrict__ captions, const float* __restrict__ emb,
                             _Float16* __restrict__ embs16) {
  const size_t i = (size_t)blockIdx.x * 256 + threadIdx.x;
  if (i >= (size_t)B_ * T_ * E_) return;
  const int e = (int)(i % E_);
  const size_t bt = i / E_;
  const int t = (int)(bt % T_);
  const int b = (int)(bt / T_);
  const int tok = captions[b * S_ + t];
  embs16[i] = (_Float16)emb[(size_t)tok * E_ + e];
}

__global__ void build_xh_kernel(const _Float16* __restrict__ embs16,
                                const float* __restrict__ gatelin,
                                const float* __restrict__ awe,
                                const _Float16* __restrict__ h16,
                                _Float16* __restrict__ xh, int t) {
  const int i = blockIdx.x * 256 + threadIdx.x;
  if (i >= B_ * KXH) return;
  const int b = i / KXH, j = i % KXH;
  _Float16 v;
  if (j < E_) {
    v = embs16[((size_t)b * T_ + t) * E_ + j];
  } else if (j < E_ + D_) {
    const int d = j - E_;
    const float g = 1.f / (1.f + __expf(-gatelin[b * D_ + d]));
    v = (_Float16)(g * awe[b * D_ + d]);
  } else {
    v = h16[b * L_ + (j - (E_ + D_))];
  }
  xh[i] = v;
}

__global__ void lstm_kernel(const float* __restrict__ gates, const int* __restrict__ lengths,
                            float* __restrict__ h, float* __restrict__ c,
                            _Float16* __restrict__ h16, int t) {
  const int i = blockIdx.x * 256 + threadIdx.x;
  if (i >= B_ * L_) return;
  const int b = i / L_, j = i % L_;
  const float* g = gates + (size_t)b * (4 * L_);
  const float ig = 1.f / (1.f + __expf(-g[j]));
  const float fg = 1.f / (1.f + __expf(-g[L_ + j]));
  const float gg = tanhf(g[2 * L_ + j]);
  const float og = 1.f / (1.f + __expf(-g[3 * L_ + j]));
  const float cn = fg * c[i] + ig * gg;
  const float hn = og * tanhf(cn);
  const bool msk = t < (lengths[b] - 1);
  const float ho = msk ? hn : h[i];
  const float co = msk ? cn : c[i];
  h[i] = ho; c[i] = co; h16[i] = (_Float16)ho;
}

// ---------------------------------------------------------------------------
extern "C" void kernel_launch(void* const* d_in, const int* in_sizes, int n_in,
                              void* d_out, int out_size, void* d_ws, size_t ws_size,
                              hipStream_t stream) {
  (void)in_sizes; (void)n_in; (void)out_size; (void)ws_size;
  const float* features = (const float*)d_in[0];
  const int*   captions = (const int*)d_in[1];
  const int*   lengths  = (const int*)d_in[2];
  const float* emb      = (const float*)d_in[3];
  const float* W_ih     = (const float*)d_in[4];
  const float* b_ih     = (const float*)d_in[5];
  const float* W_hh     = (const float*)d_in[6];
  const float* b_hh     = (const float*)d_in[7];
  const float* W_h      = (const float*)d_in[8];
  const float* b_h      = (const float*)d_in[9];
  const float* W_c      = (const float*)d_in[10];
  const float* b_c      = (const float*)d_in[11];
  const float* W_beta   = (const float*)d_in[12];
  const float* b_beta   = (const float*)d_in[13];
  const float* W_cls    = (const float*)d_in[14];
  const float* b_cls    = (const float*)d_in[15];
  const float* W_enc    = (const float*)d_in[16];
  const float* b_enc    = (const float*)d_in[17];
  const float* W_dec    = (const float*)d_in[18];
  const float* b_dec    = (const float*)d_in[19];
  const float* W_att    = (const float*)d_in[20];
  const float* b_att    = (const float*)d_in[21];

  float* Y   = (float*)d_out;                      // [B, T, V]
  float* ALP = Y + (size_t)B_ * T_ * V_;           // [B, T, P]

  // Bump-allocate workspace (~58 MB), 256B aligned slices.
  char* p = (char*)d_ws;
  auto alloc = [&](size_t bytes) -> char* {
    char* r = p; p += (bytes + 255) & ~(size_t)255; return r;
  };
  float*    att1    = (float*)alloc((size_t)B_ * P_ * A_ * 4);
  _Float16* Wenc16  = (_Float16*)alloc((size_t)A_ * D_ * 2);
  _Float16* Wdec16  = (_Float16*)alloc((size_t)A_ * L_ * 2);
  _Float16* Wbeta16 = (_Float16*)alloc((size_t)D_ * L_ * 2);
  _Float16* Wcls16  = (_Float16*)alloc((size_t)NCLS_PAD * L_ * 2);
  _Float16* Wg16    = (_Float16*)alloc((size_t)(4 * L_) * KXH * 2);
  float*    bg      = (float*)alloc((size_t)(4 * L_) * 4);
  _Float16* embs16  = (_Float16*)alloc((size_t)B_ * T_ * E_ * 2);
  float*    meanf   = (float*)alloc((size_t)B_ * D_ * 4);
  float*    hbuf    = (float*)alloc((size_t)B_ * L_ * 4);
  float*    cbuf    = (float*)alloc((size_t)B_ * L_ * 4);
  _Float16* h16     = (_Float16*)alloc((size_t)B_ * L_ * 2);
  float*    att2    = (float*)alloc((size_t)B_ * A_ * 4);
  float*    awe     = (float*)alloc((size_t)B_ * D_ * 4);
  float*    gatelin = (float*)alloc((size_t)B_ * D_ * 4);
  _Float16* xh16    = (_Float16*)alloc((size_t)B_ * KXH * 2);
  float*    gates   = (float*)alloc((size_t)B_ * (4 * L_) * 4);

  const dim3 blk(256), blkg(128);

  // ---- one-time setup (recomputed every call; deterministic) ----
  cvt_f16_kernel<<<dim3((A_ * D_ + 255) / 256), blk, 0, stream>>>(W_enc, Wenc16, (size_t)A_ * D_);
  cvt_f16_kernel<<<dim3((A_ * L_ + 255) / 256), blk, 0, stream>>>(W_dec, Wdec16, (size_t)A_ * L_);
  cvt_f16_kernel<<<dim3((D_ * L_ + 255) / 256), blk, 0, stream>>>(W_beta, Wbeta16, (size_t)D_ * L_);
  cvt_cls_kernel<<<dim3((NCLS_PAD * L_ + 255) / 256), blk, 0, stream>>>(W_cls, Wcls16);
  build_wg_kernel<<<dim3((4 * L_ * KXH + 255) / 256), blk, 0, stream>>>(W_ih, W_hh, Wg16);
  bias_sum_kernel<<<dim3((4 * L_ + 255) / 256), blk, 0, stream>>>(b_ih, b_hh, bg);
  embed_kernel<<<dim3((B_ * T_ * E_ + 255) / 256), blk, 0, stream>>>(captions, emb, embs16);
  mean_kernel<<<dim3(B_, D_ / 256), blk, 0, stream>>>(features, meanf);
  h0c0_kernel<<<dim3(B_, L_ / 256), blk, 0, stream>>>(meanf, W_h, b_h, W_c, b_c,
                                                      hbuf, cbuf, h16);
  // att1 = feats @ W_enc^T + b_enc   (26 GFLOP, fp32 A converted in-register)
  wmma_gemm_kernel<true><<<dim3(A_ / 64, (B_ * P_) / 64), blkg, 0, stream>>>(
      features, Wenc16, b_enc, att1, (_Float16*)nullptr, B_ * P_, A_, D_);

  // ---- recurrent scan ----
  for (int t = 0; t < T_; ++t) {
    // att2 = h @ W_dec^T + b_dec
    wmma_gemm_kernel<false><<<dim3(A_ / 64, 1), blkg, 0, stream>>>(
        h16, Wdec16, b_dec, att2, (_Float16*)nullptr, B_, A_, L_);
    // softmax attention + context vector (also writes alphas output, masked)
    attn_kernel<<<dim3(B_), blk, 0, stream>>>(att1, att2, W_att, b_att, features,
                                              lengths, awe, ALP, t);
    // beta-gate preactivation: h @ W_beta^T + b_beta
    wmma_gemm_kernel<false><<<dim3(D_ / 64, 1), blkg, 0, stream>>>(
        h16, Wbeta16, b_beta, gatelin, (_Float16*)nullptr, B_, D_, L_);
    // xh = [emb_t | sigmoid(gatelin)*awe | h]  (f16)
    build_xh_kernel<<<dim3((B_ * KXH + 255) / 256), blk, 0, stream>>>(
        embs16, gatelin, awe, h16, xh16, t);
    // gates = xh @ [W_ih|W_hh]^T + (b_ih + b_hh)
    wmma_gemm_kernel<false><<<dim3((4 * L_) / 64, 1), blkg, 0, stream>>>(
        xh16, Wg16, bg, gates, (_Float16*)nullptr, B_, 4 * L_, KXH);
    // LSTM pointwise + sequence mask; refresh h16 carry
    lstm_kernel<<<dim3((B_ * L_ + 255) / 256), blk, 0, stream>>>(gates, lengths,
                                                                 hbuf, cbuf, h16, t);
    // y = h @ W_cls^T + b_cls, masked, written straight into d_out
    wmma_cls_kernel<<<dim3(NCLS_PAD / 64, 1), blkg, 0, stream>>>(
        h16, Wcls16, b_cls, lengths, Y, t);
  }
}
